// RoPESelfKANtentionND_45612552683528
// MI455X (gfx1250) — matrix operand order repaired and checked by reading, not compiled, not run on hardware
//
#include <hip/hip_runtime.h>
#include <hip/hip_bf16.h>
#include <math.h>

typedef __bf16 bf16_t;
typedef __attribute__((ext_vector_type(8)))  __bf16 v8bf;
typedef __attribute__((ext_vector_type(16))) __bf16 v16bf;
typedef __attribute__((ext_vector_type(8)))  float  v8f;
typedef __attribute__((ext_vector_type(4)))  unsigned int v4u;
typedef __attribute__((ext_vector_type(8)))  int v8i;
typedef __attribute__((ext_vector_type(4)))  int v4i;

#define NB 64
#define NC 768
#define NL 197
#define ND 4
#define NCD (NC * ND)   // 3072
#define LZ  224         // padded Z length (1 halo + 197 + tail zeros)
#define LP  208         // padded token count (13 * 16)
#define KJ  224         // padded attention K dim (7 * 32)
#define NPAIR (NC / 2)  // 384 RoPE channel pairs
#define BNN (NB * NL)   // 12608 elements per channel for BN stats
#define ZT0 20          // TDM tile dim0 (l columns staged per chunk)
#define NKC (NCD / 32)  // 96 K-chunks in conv GEMM

#ifndef __has_builtin
#define __has_builtin(x) 0
#endif
#if __has_builtin(__builtin_amdgcn_tensor_load_to_lds) && \
    __has_builtin(__builtin_amdgcn_s_wait_tensorcnt)
#define HAVE_TDM 1
#else
#define HAVE_TDM 0
#endif

static __device__ __forceinline__ v8f wmma_bf16(v16bf a, v16bf b, v8f c) {
  // D = A(16x32 bf16) * B(32x16 bf16) + C(16x16 f32)
  return __builtin_amdgcn_wmma_f32_16x16x32_bf16(false, a, false, b, (short)0, c,
                                                 false, false);
}

static __device__ __forceinline__ v16bf cat8(v8bf lo, v8bf hi) {
  return __builtin_shufflevector(lo, hi, 0, 1, 2, 3, 4, 5, 6, 7,
                                 8, 9, 10, 11, 12, 13, 14, 15);
}

#if HAVE_TDM
// Issue a TDM DMA of a 2D bf16 tile (ZT0 cols x 32 rows) from Z into LDS.
// D# per CDNA5 ISA ch.8: group0 = {count, lds_addr, global_addr, type=2},
// group1 = {data_size=2B, tensor_dim0=LZ, tensor_dim1=NCD, tile 20x32,
//           tensor_dim0_stride=LZ}.
static __device__ __forceinline__ void tdm_load_tile(unsigned lds_byte,
                                                     const bf16_t* gptr) {
  unsigned long long ga = (unsigned long long)(size_t)gptr;
  v4u g0 = {1u,                               // count=1, is_restore=0
            lds_byte,                          // lds_addr (bytes)
            (unsigned)ga,                      // global_addr[31:0]
            (unsigned)((ga >> 32) & 0x01ffffffu) | (2u << 30)}; // addr[56:32]|type=2
  v8i g1 = {(int)(1u << 16),                   // data_size code 1 -> 2 bytes
            (int)((unsigned)LZ << 16),         // tensor_dim0[15:0] << 16
            (int)((unsigned)NCD << 16),        // dim0 hi=0 | tensor_dim1[15:0]<<16
            (int)((unsigned)ZT0 << 16),        // dim1 hi=0 | tile_dim0<<16
            32,                                // tile_dim1=32 | tile_dim2=0
            (int)LZ,                           // tensor_dim0_stride[31:0]
            0, 0};
  v4i zz = {0, 0, 0, 0};
#if __clang_major__ >= 23
  v8i zz8 = {0, 0, 0, 0, 0, 0, 0, 0};
  __builtin_amdgcn_tensor_load_to_lds(g0, g1, zz, zz, zz8, 0);
#else
  __builtin_amdgcn_tensor_load_to_lds(g0, g1, zz, zz, 0);
#endif
}
#endif

// ---------------------------------------------------------------- zero fill
__global__ void k_zero_f4(float4* __restrict__ p, long long n) {
  long long i = (long long)blockIdx.x * blockDim.x + threadIdx.x;
  if (i < n) p[i] = make_float4(0.f, 0.f, 0.f, 0.f);
}

// ------------------------------------------------- KACN Chebyshev basis -> Z
__global__ void k_kacn_basis(const float* __restrict__ x, bf16_t* __restrict__ Z) {
  long long i = (long long)blockIdx.x * blockDim.x + threadIdx.x;
  if (i >= (long long)NB * NC * NL) return;
  int l = (int)(i % NL);
  long long t = i / NL;
  int ci = (int)(t % NC);
  int b  = (int)(t / NC);
  float ac = acosf(tanhf(x[i]));
  size_t base = ((size_t)b * NCD + (size_t)ci * ND) * LZ + 1 + l;
#pragma unroll
  for (int d = 0; d < ND; ++d)
    Z[base + (size_t)d * LZ] = (bf16_t)cosf((float)d * ac);
}

// ------------------------ repack W (Cout, Cin*4, 3) f32 -> Wb[tap][co][j] bf16
__global__ void k_wcvt(const float* __restrict__ W, bf16_t* __restrict__ Wb) {
  long long i = (long long)blockIdx.x * blockDim.x + threadIdx.x;
  if (i >= (long long)NC * NCD * 3) return;
  int tap = (int)(i % 3);
  long long t = i / 3;
  int j  = (int)(t % NCD);
  int co = (int)(t / NCD);
  Wb[((size_t)tap * NC + co) * NCD + j] = (bf16_t)W[i];
}

// ------------------------------------- conv1d(k=3,pad=1) as 3 shifted GEMMs
// Y[b][co][l] = sum_tap sum_j Wb[tap][co][j] * Z[b][j][l + tap]  (halo layout)
// One wave per 16x16 tile. B panel staged in LDS by the Tensor Data Mover
// (double buffered, s_wait_tensorcnt), A fragments are two b128 loads.
__global__ void k_kacn_conv_wmma(const bf16_t* __restrict__ Wb,
                                 const bf16_t* __restrict__ Z,
                                 float* __restrict__ Y) {
  __shared__ __align__(16) bf16_t zbuf[2][32 * ZT0];
  int lane = threadIdx.x;
  int hi = lane >> 4, mr = lane & 15;
  int ib = blockIdx.x * 16;   // token tile base
  int cb = blockIdx.y * 16;   // cout tile base
  int b  = blockIdx.z;
  const bf16_t* Zb = Z + (size_t)b * NCD * LZ;
  v8f acc = {};
#if HAVE_TDM
  tdm_load_tile(0u, Zb + /*kbase=0*/ (size_t)0 * LZ + ib);
#endif
  for (int kc = 0; kc < NKC; ++kc) {
    int kbase = kc * 32;
    int buf = kc & 1;
#if HAVE_TDM
    if (kc + 1 < NKC) {
      tdm_load_tile((unsigned)((buf ^ 1) * 32 * ZT0 * 2),
                    Zb + (size_t)(kbase + 32) * LZ + ib);
      __builtin_amdgcn_s_wait_tensorcnt((short)1);
    } else {
      __builtin_amdgcn_s_wait_tensorcnt((short)0);
    }
#else
    // fallback: single-wave cooperative stage (DS in-order within wave)
    for (int t = 0; t < ZT0; ++t)
      zbuf[buf][lane * ZT0 + t] = Zb[(size_t)(kbase + lane) * LZ + ib + t];
#endif
#pragma unroll
    for (int tap = 0; tap < 3; ++tap) {
      const bf16_t* Wrow = Wb + ((size_t)tap * NC + cb + mr) * NCD + kbase + hi * 8;
      v16bf a = cat8(*(const v8bf*)Wrow, *(const v8bf*)(Wrow + 16));
      v16bf bm;
#pragma unroll
      for (int n = 0; n < 16; ++n) bm[n] = zbuf[buf][((n & 7) + hi * 8 + (n >> 3) * 16) * ZT0 + tap + mr];
      acc = wmma_bf16(a, bm, acc);
    }
  }
  int col = ib + mr;
  if (col < NL) {
#pragma unroll
    for (int v = 0; v < 8; ++v)
      Y[((size_t)b * NC + (cb + hi * 8 + v)) * NL + col] = acc[v];
  }
}

// --------------------------------------------- per-channel BN stats (train)
__global__ void k_bn_stats(const float* __restrict__ Y,
                           float* __restrict__ mean_out,
                           float* __restrict__ rstd_out) {
  __shared__ float s1[256], s2[256];
  int c = blockIdx.x, tid = threadIdx.x;
  float s = 0.f, q = 0.f;
  for (int i = tid; i < BNN; i += 256) {
    int b = i / NL, l = i % NL;
    float v = Y[((size_t)b * NC + c) * NL + l];
    s += v; q += v * v;
  }
  s1[tid] = s; s2[tid] = q; __syncthreads();
  for (int st = 128; st > 0; st >>= 1) {
    if (tid < st) { s1[tid] += s1[tid + st]; s2[tid] += s2[tid + st]; }
    __syncthreads();
  }
  if (tid == 0) {
    float m = s1[0] / (float)BNN;
    float var = s2[0] / (float)BNN - m * m;
    mean_out[c] = m;
    rstd_out[c] = rsqrtf(var + 1e-5f);
  }
}

// --------------------- BN apply + transpose + 2D mixed RoPE -> Qr, Kr (bf16)
__global__ void k_bn_rope(const float* __restrict__ Yq, const float* __restrict__ Yk,
                          const float* __restrict__ stats,
                          const float* __restrict__ qs, const float* __restrict__ qb,
                          const float* __restrict__ ks, const float* __restrict__ kb,
                          const float* __restrict__ freqs,
                          bf16_t* __restrict__ Qr, bf16_t* __restrict__ Kr) {
  int n = blockIdx.x, b = blockIdx.y, p = threadIdx.x;  // p < 384
  int c0 = 2 * p, c1 = c0 + 1;
  size_t o = ((size_t)b * LP + n) * NC + c0;
  if (n >= NL) {  // zero pad rows so padded GEMM tiles are clean
    Qr[o] = (bf16_t)0.f; Qr[o + 1] = (bf16_t)0.f;
    Kr[o] = (bf16_t)0.f; Kr[o + 1] = (bf16_t)0.f;
    return;
  }
  size_t i0 = ((size_t)b * NC + c0) * NL + n;
  size_t i1 = ((size_t)b * NC + c1) * NL + n;
  float q0 = (Yq[i0] - stats[c0]) * stats[NC + c0] * qs[c0] + qb[c0];
  float q1 = (Yq[i1] - stats[c1]) * stats[NC + c1] * qs[c1] + qb[c1];
  float k0 = (Yk[i0] - stats[2 * NC + c0]) * stats[3 * NC + c0] * ks[c0] + kb[c0];
  float k1 = (Yk[i1] - stats[2 * NC + c1]) * stats[3 * NC + c1] * ks[c1] + kb[c1];
  if (n > 0) {  // CLS token (n==0) is not rotated
    float t = (float)(n - 1);
    float tx = fmodf(t, 14.0f);
    float ty = floorf(t / 14.0f);
    float ang = tx * freqs[p] + ty * freqs[NPAIR + p];
    float sa, ca;
    sincosf(ang, &sa, &ca);
    float r0 = q0 * ca - q1 * sa, r1 = q0 * sa + q1 * ca;
    q0 = r0; q1 = r1;
    r0 = k0 * ca - k1 * sa; r1 = k0 * sa + k1 * ca;
    k0 = r0; k1 = r1;
  }
  Qr[o] = (bf16_t)q0; Qr[o + 1] = (bf16_t)q1;
  Kr[o] = (bf16_t)k0; Kr[o + 1] = (bf16_t)k1;
}

// --------------------------------------- BN apply on V -> bf16 (B, 768, 224)
__global__ void k_bn_v(const float* __restrict__ Yv, const float* __restrict__ statsv,
                       const float* __restrict__ vs, const float* __restrict__ vb,
                       bf16_t* __restrict__ Vb) {
  int l = threadIdx.x;         // 0..223
  int c = blockIdx.x, b = blockIdx.y;
  float val = 0.f;
  if (l < NL)
    val = (Yv[((size_t)b * NC + c) * NL + l] - statsv[c]) * statsv[NC + c] * vs[c] + vb[c];
  Vb[((size_t)b * NC + c) * KJ + l] = (bf16_t)val;
}

// ------------------------------------------------ energy = Qr * Kr^T (WMMA)
__global__ void k_energy_wmma(const bf16_t* __restrict__ Qr,
                              const bf16_t* __restrict__ Kr,
                              float* __restrict__ E) {
  int lane = threadIdx.x;
  int hi = lane >> 4, mr = lane & 15;
  int jb = blockIdx.x * 16, ibs = blockIdx.y * 16, b = blockIdx.z;
  const bf16_t* Qrow = Qr + ((size_t)b * LP + ibs + mr) * NC;
  const bf16_t* Kb   = Kr + (size_t)b * LP * NC;
  v8f acc = {};
  for (int kc = 0; kc < NC / 32; ++kc) {
    int kbase = kc * 32;
    v16bf a = cat8(*(const v8bf*)(Qrow + kbase + hi * 8),
                   *(const v8bf*)(Qrow + kbase + 16 + hi * 8));
    __builtin_prefetch(Kb + (size_t)(jb + (lane & 15)) * NC + kbase + 64, 0, 1);
    v16bf bm;
#pragma unroll
    for (int n = 0; n < 16; ++n)
      bm[n] = Kb[(size_t)(jb + mr) * NC + kbase + (n & 7) + hi * 8 + (n >> 3) * 16];
    acc = wmma_bf16(a, bm, acc);
  }
#pragma unroll
  for (int v = 0; v < 8; ++v)
    E[((size_t)b * LP + ibs + hi * 8 + v) * LP + jb + mr] = acc[v];
}

// ------------------------------------------- row softmax -> bf16 att (padded)
__global__ void k_softmax(const float* __restrict__ E, bf16_t* __restrict__ A) {
  __shared__ float red[256];
  int i = blockIdx.x, b = blockIdx.y, tid = threadIdx.x;
  size_t arow = ((size_t)b * LP + i) * KJ;
  if (i >= NL) {  // pad rows: all zero
    for (int j = tid; j < KJ; j += 256) A[arow + j] = (bf16_t)0.f;
    return;
  }
  size_t erow = ((size_t)b * LP + i) * LP;
  float v = (tid < NL) ? E[erow + tid] : -INFINITY;
  red[tid] = v; __syncthreads();
  for (int st = 128; st > 0; st >>= 1) {
    if (tid < st) red[tid] = fmaxf(red[tid], red[tid + st]);
    __syncthreads();
  }
  float mx = red[0]; __syncthreads();
  float e = (tid < NL) ? __expf(v - mx) : 0.f;
  red[tid] = e; __syncthreads();
  for (int st = 128; st > 0; st >>= 1) {
    if (tid < st) red[tid] += red[tid + st];
    __syncthreads();
  }
  float inv = 1.f / red[0];
  if (tid < KJ) A[arow + tid] = (bf16_t)((tid < NL) ? e * inv : 0.f);
}

// -------------------------------------------- out = V * att^T (WMMA, K = j)
__global__ void k_out_wmma(const bf16_t* __restrict__ Vb,
                           const bf16_t* __restrict__ A,
                           float* __restrict__ O) {
  int lane = threadIdx.x;
  int hi = lane >> 4, mr = lane & 15;
  int ibs = blockIdx.x * 16, cb = blockIdx.y * 16, b = blockIdx.z;
  const bf16_t* Vrow = Vb + ((size_t)b * NC + cb + mr) * KJ;
  const bf16_t* Ab   = A + (size_t)b * LP * KJ;
  v8f acc = {};
#pragma unroll
  for (int kc = 0; kc < KJ / 32; ++kc) {
    int kbase = kc * 32;
    v16bf a = cat8(*(const v8bf*)(Vrow + kbase + hi * 8),
                   *(const v8bf*)(Vrow + kbase + 16 + hi * 8));
    v16bf bm;
#pragma unroll
    for (int n = 0; n < 16; ++n)
      bm[n] = Ab[(size_t)(ibs + mr) * KJ + kbase + (n & 7) + hi * 8 + (n >> 3) * 16];
    acc = wmma_bf16(a, bm, acc);
  }
  int col = ibs + mr;
  if (col < NL) {
#pragma unroll
    for (int v = 0; v < 8; ++v)
      O[((size_t)b * NC + (cb + hi * 8 + v)) * NL + col] = acc[v];
  }
}

// ------------------------------- final BN(gamma*x + out) -> d_out, two pass
__global__ void k_final_bn(const float* __restrict__ x, const float* __restrict__ O,
                           const float* __restrict__ gamma,
                           const float* __restrict__ ns, const float* __restrict__ nbias,
                           float* __restrict__ outp) {
  __shared__ float s1[256], s2[256];
  int c = blockIdx.x, tid = threadIdx.x;
  float g = gamma[0];
  float s = 0.f, q = 0.f;
  for (int i = tid; i < BNN; i += 256) {
    int b = i / NL, l = i % NL;
    size_t idx = ((size_t)b * NC + c) * NL + l;
    float r = g * x[idx] + O[idx];
    s += r; q += r * r;
  }
  s1[tid] = s; s2[tid] = q; __syncthreads();
  for (int st = 128; st > 0; st >>= 1) {
    if (tid < st) { s1[tid] += s1[tid + st]; s2[tid] += s2[tid + st]; }
    __syncthreads();
  }
  if (tid == 0) {
    float m = s1[0] / (float)BNN;
    float var = s2[0] / (float)BNN - m * m;
    s1[0] = m;
    s2[0] = rsqrtf(var + 1e-5f);
  }
  __syncthreads();
  float m = s1[0], rs = s2[0];
  float sc = ns[c], bi = nbias[c];
  for (int i = tid; i < BNN; i += 256) {
    int b = i / NL, l = i % NL;
    size_t idx = ((size_t)b * NC + c) * NL + l;
    float r = g * x[idx] + O[idx];
    outp[idx] = (r - m) * rs * sc + bi;
  }
}

extern "C" void kernel_launch(void* const* d_in, const int* in_sizes, int n_in,
                              void* d_out, int out_size, void* d_ws, size_t ws_size,
                              hipStream_t stream) {
  const float* x     = (const float*)d_in[0];
  const float* Wq    = (const float*)d_in[1];
  const float* Wk    = (const float*)d_in[2];
  const float* Wv    = (const float*)d_in[3];
  const float* bnq_s = (const float*)d_in[4];
  const float* bnq_b = (const float*)d_in[5];
  const float* bnk_s = (const float*)d_in[6];
  const float* bnk_b = (const float*)d_in[7];
  const float* bnv_s = (const float*)d_in[8];
  const float* bnv_b = (const float*)d_in[9];
  const float* gamma = (const float*)d_in[10];
  const float* freqs = (const float*)d_in[11];
  const float* nrm_s = (const float*)d_in[12];
  const float* nrm_b = (const float*)d_in[13];

  // ---- workspace layout (256B-aligned chunks) ----
  const size_t ZB  = (size_t)NB * NCD * LZ * sizeof(bf16_t);  // 88.1 MB
  const size_t YB  = (size_t)NB * NC * NL * sizeof(float);    // 38.7 MB
  const size_t QRB = (size_t)NB * LP * NC * sizeof(bf16_t);   // 20.4 MB
  const size_t VBB = (size_t)NB * NC * KJ * sizeof(bf16_t);   // 22.0 MB
  const size_t WBB = (size_t)3 * NC * NCD * sizeof(bf16_t);   // 14.2 MB
  const size_t STB = (size_t)8 * NC * sizeof(float);

  char* w = (char*)d_ws;
  size_t off = 0;
  auto take = [&](size_t bytes) { size_t o = off; off += (bytes + 255) & ~(size_t)255; return o; };
  bf16_t* Z   = (bf16_t*)(w + take(ZB));
  float*  Yq  = (float*)(w + take(YB));
  float*  Yk  = (float*)(w + take(YB));
  float*  Yv  = (float*)(w + take(YB));
  bf16_t* Qr  = (bf16_t*)(w + take(QRB));
  bf16_t* Kr  = (bf16_t*)(w + take(QRB));
  bf16_t* Vb  = (bf16_t*)(w + take(VBB));
  bf16_t* Wb  = (bf16_t*)(w + take(WBB));
  float*  st  = (float*)(w + take(STB));
  (void)ws_size; (void)n_in; (void)in_sizes; (void)out_size;
  // dead-buffer aliases (lifetimes do not overlap):
  float*  E   = (float*)Yq;          // energy (B,208,208) f32, live after rope
  bf16_t* Att = (bf16_t*)Yk;         // att (B,208,224) bf16, live after rope
  float*  O   = (float*)Z;           // attention output (B,768,197) f32, live after conv

  // 1) zero Z (halo + tail padding), basis fill, weight repack
  {
    long long n4 = (long long)(ZB / 16);
    k_zero_f4<<<dim3((unsigned)((n4 + 255) / 256)), dim3(256), 0, stream>>>((float4*)Z, n4);
    long long tot = (long long)NB * NC * NL;
    k_kacn_basis<<<dim3((unsigned)((tot + 255) / 256)), dim3(256), 0, stream>>>(x, Z);
    long long wt = (long long)NC * NCD * 3;
    k_wcvt<<<dim3((unsigned)((wt + 255) / 256)), dim3(256), 0, stream>>>(Wq, Wb);
  }
  // 2) three conv GEMMs (WMMA + TDM-staged B panels)
  {
    dim3 g(13, NC / 16, NB), blk(32);
    k_kacn_conv_wmma<<<g, blk, 0, stream>>>(Wb, Z, Yq);
    long long wt = (long long)NC * NCD * 3;
    k_wcvt<<<dim3((unsigned)((wt + 255) / 256)), dim3(256), 0, stream>>>(Wk, Wb);
    k_kacn_conv_wmma<<<g, blk, 0, stream>>>(Wb, Z, Yk);
    k_wcvt<<<dim3((unsigned)((wt + 255) / 256)), dim3(256), 0, stream>>>(Wv, Wb);
    k_kacn_conv_wmma<<<g, blk, 0, stream>>>(Wb, Z, Yv);
  }
  // 3) BN stats: [meanq|rstdq|meank|rstdk|meanv|rstdv]
  k_bn_stats<<<dim3(NC), dim3(256), 0, stream>>>(Yq, st + 0 * NC, st + 1 * NC);
  k_bn_stats<<<dim3(NC), dim3(256), 0, stream>>>(Yk, st + 2 * NC, st + 3 * NC);
  k_bn_stats<<<dim3(NC), dim3(256), 0, stream>>>(Yv, st + 4 * NC, st + 5 * NC);
  // 4) BN+transpose+RoPE -> Qr/Kr ; BN -> Vb
  k_bn_rope<<<dim3(LP, NB), dim3(NPAIR), 0, stream>>>(Yq, Yk, st, bnq_s, bnq_b,
                                                      bnk_s, bnk_b, freqs, Qr, Kr);
  k_bn_v<<<dim3(NC, NB), dim3(KJ), 0, stream>>>(Yv, st + 4 * NC, bnv_s, bnv_b, Vb);
  // 5) attention
  k_energy_wmma<<<dim3(13, 13, NB), dim3(32), 0, stream>>>(Qr, Kr, E);
  k_softmax<<<dim3(LP, NB), dim3(256), 0, stream>>>(E, Att);
  k_out_wmma<<<dim3(13, NC / 16, NB), dim3(32), 0, stream>>>(Vb, Att, O);
  // 6) final BN(gamma*x + out)
  k_final_bn<<<dim3(NC), dim3(256), 0, stream>>>(x, O, gamma, nrm_s, nrm_b,
                                                 (float*)d_out);
}
